// ODEFunc_65747359367616
// MI455X (gfx1250) — matrix operand order repaired and checked by reading, compiled, not run on hardware
//
#include <hip/hip_runtime.h>

// ---------------------------------------------------------------------------
// MI455X (gfx1250): relu -> conv3x3(256->256) -> avgpool(8x8) -> pruned MLP
//                   -> NN-upsample -> instance-norm
// bf16 WMMA everywhere; K-split GEMMs with f32 atomic reduction; non-temporal
// streaming of the 256MB weight tensors with register double-buffering;
// async global->LDS staging (ASYNCcnt) for bf16 tiles where available.
// ---------------------------------------------------------------------------

typedef __bf16 bf16;
typedef __attribute__((ext_vector_type(16))) __bf16 bf16x16;
typedef __attribute__((ext_vector_type(8)))  __bf16 bf16x8;
typedef __attribute__((ext_vector_type(8)))  float  v8f;
typedef __attribute__((ext_vector_type(4)))  float  v4f;
typedef int v4i __attribute__((vector_size(16)));   // matches builtin param

#define Bn     32
#define Cn     256
#define Hn     32
#define Wn     32
#define NNODE  16384      // C * 8 * 8
#define FEAT   2048
#define KCONV  2304       // Cin * 3 * 3
#define NBIN   2048

// -------- CDNA5 async global->LDS copies (GLOBAL_LOAD_ASYNC_TO_LDS_B128) ---
#if defined(__has_builtin)
#if __has_builtin(__builtin_amdgcn_global_load_async_to_lds_b128) && \
    __has_builtin(__builtin_amdgcn_s_wait_asynccnt)
#define HAVE_ASYNC_LDS 1
#endif
#endif
#ifndef HAVE_ASYNC_LDS
#define HAVE_ASYNC_LDS 0
#endif

// copy 16B from global to LDS; per-lane addresses; tracked by ASYNCcnt
static __device__ __forceinline__ void g2l_b128(const void* gp, void* lp) {
#if HAVE_ASYNC_LDS
    __builtin_amdgcn_global_load_async_to_lds_b128(
        (__attribute__((address_space(1))) v4i*)(unsigned long long)gp,
        (__attribute__((address_space(3))) v4i*)(unsigned)(unsigned long long)lp,
        0, 0);
#else
    *(bf16x8*)lp = *(const bf16x8*)gp;
#endif
}

static __device__ __forceinline__ void wait_async_lds() {
#if HAVE_ASYNC_LDS
    __builtin_amdgcn_s_wait_asynccnt(0);
#endif
}

static __device__ __forceinline__ bf16 f2bf(float f) {
    unsigned u = __builtin_bit_cast(unsigned, f);
    unsigned r = u + 0x7FFFu + ((u >> 16) & 1u);   // round-to-nearest-even
    unsigned short s = (unsigned short)(r >> 16);
    return __builtin_bit_cast(bf16, s);
}

static __device__ __forceinline__ v8f vzero() {
    v8f z = {0.f,0.f,0.f,0.f,0.f,0.f,0.f,0.f};
    return z;
}

// ---------------------------------------------------------------- utilities
__global__ void k_zero_u32(unsigned* __restrict__ p, long long n) {
    long long i = (long long)blockIdx.x * blockDim.x + threadIdx.x;
    long long s = (long long)gridDim.x * blockDim.x;
    for (; i < n; i += s) p[i] = 0u;
}

__global__ void k_relu_cast(const float* __restrict__ x, bf16* __restrict__ xb,
                            long long n) {
    long long i = (long long)blockIdx.x * blockDim.x + threadIdx.x;
    long long s = (long long)gridDim.x * blockDim.x;
    for (; i < n; i += s) {
        float v = x[i];
        xb[i] = f2bf(v > 0.f ? v : 0.f);
    }
}

// conv_w [O=256][K=2304] -> wb [K=2304][O=256]  (K-major for B-tile streaming)
__global__ void k_cast_w(const float* __restrict__ cw, bf16* __restrict__ wb) {
    long long i = (long long)blockIdx.x * blockDim.x + threadIdx.x;
    if (i >= (long long)Cn * KCONV) return;
    int o = (int)(i / KCONV);
    int k = (int)(i - (long long)o * KCONV);
    wb[(long long)k * Cn + o] = f2bf(cw[i]);
}

// ---------------------------------------------- conv as implicit GEMM (WMMA)
// M = B*H*W = 32768 pixels, N = 256 out-channels, K = 2304.
// 128x128 tile/block, 8 waves; epilogue fuses 4x4 avg-pool via atomicAdd.
__global__ __launch_bounds__(256)
void k_conv_wmma(const bf16* __restrict__ xb, const bf16* __restrict__ wb,
                 float* __restrict__ pooled) {
    __shared__ __align__(32) bf16 As[128 * 32];
    __shared__ __align__(32) bf16 Bs[32 * 128];

    const int m0 = blockIdx.x * 128;
    const int n0 = blockIdx.y * 128;
    const int t = threadIdx.x;
    const int lane = t & 31, wv = t >> 5;

    v8f acc[8];
#pragma unroll
    for (int j = 0; j < 8; ++j) acc[j] = vzero();

    const int ar = t >> 1;              // A-stage row   0..127
    const int akh = (t & 1) * 16;       // A-stage k-half
    const int am = m0 + ar;
    const int ab = am >> 10;
    const int ah = (am >> 5) & 31;
    const int aw = am & 31;

    for (int k0 = 0; k0 < KCONV; k0 += 32) {
        // ---- stage B tile 32x128 bf16 via async DMA (weights L2-resident)
        {
            int kr = t >> 3;            // 0..31
            int nc = (t & 7) * 16;      // 0..112
            const bf16* src = &wb[(long long)(k0 + kr) * Cn + n0 + nc];
            __builtin_prefetch(src + 32LL * Cn, 0, 1);   // next K block
            g2l_b128(src,     &Bs[kr * 128 + nc]);
            g2l_b128(src + 8, &Bs[kr * 128 + nc + 8]);
        }
        // ---- stage A tile (im2col gather), 128x32 bf16
#pragma unroll
        for (int kk = 0; kk < 16; ++kk) {
            int k = k0 + akh + kk;
            int cin = k / 9, rem = k - cin * 9;
            int kh = rem / 3, kw = rem - kh * 3;
            int ih = ah + kh - 1, iw = aw + kw - 1;
            bf16 v = f2bf(0.f);
            if (ih >= 0 && ih < Hn && iw >= 0 && iw < Wn)
                v = xb[(((long long)ab * Cn + cin) * Hn + ih) * Wn + iw];
            As[ar * 32 + akh + kk] = v;
        }
        wait_async_lds();
        __syncthreads();

        bf16x16 a = *(const bf16x16*)&As[(wv * 16 + (lane & 15)) * 32 +
                                         (lane >> 4) * 16];
#pragma unroll
        for (int j = 0; j < 8; ++j) {
            bf16x16 bf;
#pragma unroll
            for (int s = 0; s < 16; ++s)
                bf[s] = Bs[((lane >> 4) * 16 + s) * 128 + j * 16 + (lane & 15)];
            acc[j] = __builtin_amdgcn_wmma_f32_16x16x32_bf16(
                false, a, false, bf, (short)0, acc[j], false, false);
        }
        __syncthreads();
    }

    // ---- epilogue: fused 4x4 avg-pool accumulation (sum; /16 applied later)
    const int Mh = (lane >> 4) * 8;
    const int Nl = lane & 15;
#pragma unroll
    for (int j = 0; j < 8; ++j) {
#pragma unroll
        for (int r = 0; r < 8; ++r) {
            int m = m0 + wv * 16 + Mh + r;
            int n = n0 + j * 16 + Nl;
            int b = m >> 10, hh = (m >> 5) & 31, ww = m & 31;
            atomicAdd(&pooled[(((long long)b * Cn + n) * 8 + (hh >> 2)) * 8 +
                              (ww >> 2)],
                      acc[j][r]);
        }
    }
}

// pooled sums -> bf16 nodes (mean = sum/16); layouts coincide flat.
__global__ void k_pool_to_nodes(const float* __restrict__ pooled,
                                bf16* __restrict__ nodes, long long n) {
    long long i = (long long)blockIdx.x * blockDim.x + threadIdx.x;
    long long s = (long long)gridDim.x * blockDim.x;
    for (; i < n; i += s) nodes[i] = f2bf(pooled[i] * (1.f / 16.f));
}

// ------------------------------------------------ prune threshold (90th pct)
// LDS-privatized histogram; NT vector loads (weights must not pollute L2).
__global__ __launch_bounds__(256)
void k_hist(const float* __restrict__ w, long long n4, float invbin,
            int* __restrict__ hist) {
    __shared__ int lh[NBIN];
    const int t = threadIdx.x;
    for (int i = t; i < NBIN; i += 256) lh[i] = 0;
    __syncthreads();

    long long i = (long long)blockIdx.x * blockDim.x + t;
    long long s = (long long)gridDim.x * blockDim.x;
    for (; i < n4; i += s) {
        v4f v = __builtin_nontemporal_load((const v4f*)w + i);
#pragma unroll
        for (int e = 0; e < 4; ++e) {
            float a = fabsf(v[e]);
            int b = (int)(a * invbin);
            if (b > NBIN - 1) b = NBIN - 1;
            atomicAdd(&lh[b], 1);
        }
    }
    __syncthreads();
    for (int i2 = t; i2 < NBIN; i2 += 256)
        if (lh[i2]) atomicAdd(&hist[i2], lh[i2]);
}

__global__ void k_thr(const int* __restrict__ hist, float binw, long long total,
                      float* __restrict__ thr) {
    if (threadIdx.x == 0 && blockIdx.x == 0) {
        long long target = (long long)(0.9 * (double)total);
        long long cum = 0;
        int b = 0;
        for (; b < NBIN; ++b) { cum += hist[b]; if (cum >= target) break; }
        *thr = (float)(b + 1) * binw;
    }
}

// -------------------------------------------------------------- GEMM core
// C_partial = A[32 x Ktot] (bf16) x mask(W)[N x Ktot]^T, K-split across
// blockIdx.y, BK=64 LDS stages, f32 atomicAdd reduction into acc buffer.
// Weights: NT 16B loads double-buffered in registers (next slab loads while
// WMMAs run); A tile staged via async global->LDS DMA.
template <int KTOT, int KSPLIT>
static __device__ __forceinline__ void gemm_ksplit(
    const bf16* __restrict__ A, const float* __restrict__ W,
    float th, float* __restrict__ accbuf, int n0, int kbase) {
    __shared__ __align__(32) bf16 As[32 * 64];
    __shared__ __align__(32) bf16 Bs[64 * 128];

    const int t = threadIdx.x, lane = t & 31, wv = t >> 5;
    v8f acc0 = vzero(), acc1 = vzero();

    const int nl = t >> 1, kb = (t & 1) * 32;      // weight-stage mapping
    const int arow = t >> 3, acol = (t & 7) * 8;   // A-stage mapping

    v4f wreg[8];
    {
        const float* src = &W[(long long)(n0 + nl) * KTOT + kbase + kb];
#pragma unroll
        for (int q = 0; q < 8; ++q)
            wreg[q] = __builtin_nontemporal_load((const v4f*)(src + q * 4));
    }

    for (int k0 = kbase; k0 < kbase + KSPLIT; k0 += 64) {
        // A tile 32x64: async DMA straight into LDS
        g2l_b128(&A[(long long)arow * KTOT + k0 + acol], &As[arow * 64 + acol]);
        // B tile 64x128: mask + bf16 convert from double-buffered registers
#pragma unroll
        for (int q = 0; q < 8; ++q) {
#pragma unroll
            for (int e = 0; e < 4; ++e) {
                float w = wreg[q][e];
                Bs[(kb + q * 4 + e) * 128 + nl] = f2bf(fabsf(w) >= th ? w : 0.f);
            }
        }
        wait_async_lds();
        __syncthreads();

        // prefetch next weight slab while WMMAs execute
        if (k0 + 64 < kbase + KSPLIT) {
            const float* src = &W[(long long)(n0 + nl) * KTOT + (k0 + 64) + kb];
#pragma unroll
            for (int q = 0; q < 8; ++q)
                wreg[q] = __builtin_nontemporal_load((const v4f*)(src + q * 4));
        }

#pragma unroll
        for (int kh = 0; kh < 64; kh += 32) {
            bf16x16 a0 = *(const bf16x16*)&As[(lane & 15) * 64 + kh +
                                              (lane >> 4) * 16];
            bf16x16 a1 = *(const bf16x16*)&As[(16 + (lane & 15)) * 64 + kh +
                                              (lane >> 4) * 16];
            bf16x16 bf;
#pragma unroll
            for (int s = 0; s < 16; ++s)
                bf[s] = Bs[(kh + (lane >> 4) * 16 + s) * 128 + wv * 16 + (lane & 15)];
            acc0 = __builtin_amdgcn_wmma_f32_16x16x32_bf16(
                false, a0, false, bf, (short)0, acc0, false, false);
            acc1 = __builtin_amdgcn_wmma_f32_16x16x32_bf16(
                false, a1, false, bf, (short)0, acc1, false, false);
        }
        __syncthreads();
    }

    // K-split reduction into f32 accumulator (resident in 192MB L2)
    const int Mh = (lane >> 4) * 8;
    const int col = n0 + wv * 16 + (lane & 15);
    const int NOUT = (KTOT == NNODE) ? FEAT : NNODE;   // output width
#pragma unroll
    for (int r = 0; r < 8; ++r) {
        atomicAdd(&accbuf[(long long)(Mh + r) * NOUT + col], acc0[r]);
        atomicAdd(&accbuf[(long long)(16 + Mh + r) * NOUT + col], acc1[r]);
    }
}

// GEMM1: [32x16384] x [2048x16384]^T, 32-way K-split (512 K each)
__global__ __launch_bounds__(256)
void k_gemm1(const bf16* __restrict__ nodes, const float* __restrict__ w1,
             const float* __restrict__ thr, float* __restrict__ h_acc) {
    gemm_ksplit<NNODE, 512>(nodes, w1, thr[0], h_acc,
                            blockIdx.x * 128, blockIdx.y * 512);
}

// GEMM2: [32x2048] x [16384x2048]^T, 8-way K-split (256 K each)
__global__ __launch_bounds__(256)
void k_gemm2(const bf16* __restrict__ h, const float* __restrict__ w2,
             const float* __restrict__ thr, float* __restrict__ outs) {
    gemm_ksplit<FEAT, 256>(h, w2, thr[0], outs,
                           blockIdx.x * 128, blockIdx.y * 256);
}

// relu + bf16 cast of the GEMM1 accumulator
__global__ void k_relu_cast_h(const float* __restrict__ h_acc,
                              bf16* __restrict__ h, long long n) {
    long long i = (long long)blockIdx.x * blockDim.x + threadIdx.x;
    long long s = (long long)gridDim.x * blockDim.x;
    for (; i < n; i += s) {
        float v = h_acc[i];
        h[i] = f2bf(v > 0.f ? v : 0.f);
    }
}

// ---------------------------- instance-norm on 8x8 stats + fused NN-upsample
// Upsample repeats each value 16x, so mean/var over 32x32 == over 8x8.
__global__ __launch_bounds__(256)
void k_norm_up(const float* __restrict__ s, const float* __restrict__ gamma,
               const float* __restrict__ beta, float* __restrict__ out) {
    const int bc = blockIdx.x;           // B*C = 8192
    const int b = bc >> 8, c = bc & 255;
    const float* cell = &s[(long long)b * NNODE + c * 64];

    __shared__ float vals[64];
    __shared__ float red[2];
    const int t = threadIdx.x;
    if (t < 64) vals[t] = cell[t];
    __syncthreads();
    if (t == 0) {
        float su = 0.f, sq = 0.f;
#pragma unroll
        for (int i = 0; i < 64; ++i) { float v = vals[i]; su += v; sq += v * v; }
        float mu = su * (1.f / 64.f);
        float var = sq * (1.f / 64.f) - mu * mu;
        red[0] = mu;
        red[1] = rsqrtf(var + 1e-5f);
    }
    __syncthreads();
    const float mu = red[0], rs = red[1];
    const float g = gamma[c], be = beta[c];
    for (int i = t; i < 1024; i += 256) {
        int hh = i >> 5, ww = i & 31;
        float v = vals[(hh >> 2) * 8 + (ww >> 2)];
        out[(((long long)b * Cn + c) * Hn + hh) * Wn + ww] = (v - mu) * rs * g + be;
    }
}

// ---------------------------------------------------------------------------
extern "C" void kernel_launch(void* const* d_in, const int* in_sizes, int n_in,
                              void* d_out, int out_size, void* d_ws, size_t ws_size,
                              hipStream_t stream) {
    const float* x      = (const float*)d_in[0];
    const float* conv_w = (const float*)d_in[1];
    const float* w1     = (const float*)d_in[2];
    const float* w2     = (const float*)d_in[3];
    const float* gamma  = (const float*)d_in[4];
    const float* beta   = (const float*)d_in[5];
    float* out = (float*)d_out;

    // workspace carve-up (~23.6 MB)
    char* ws = (char*)d_ws;
    bf16*  xb       = (bf16*)(ws);                         // 16 MB
    bf16*  wb       = (bf16*)(ws + 16777216);              // 1.125 MB
    float* pooled   = (float*)(ws + 17956864);             // 2 MB
    bf16*  nodes    = (bf16*)(ws + 20054016);              // 1 MB
    float* h_acc    = (float*)(ws + 21102592);             // 256 KB
    bf16*  hbuf     = (bf16*)(ws + 21364736);              // 128 KB
    float* out_small= (float*)(ws + 21495808);             // 2 MB
    int*   hist1    = (int*)(ws + 23592960);               // 8 KB
    int*   hist2    = hist1 + NBIN;                        // 8 KB
    float* thr1     = (float*)(ws + 23609344);
    float* thr2     = thr1 + 1;

    const long long nAct  = (long long)Bn * Cn * Hn * Wn;  // 8,388,608
    const long long nPool = (long long)Bn * NNODE;         // 524,288
    const long long nW    = (long long)FEAT * NNODE;       // 33,554,432
    const long long nH    = (long long)Bn * FEAT;          // 65,536

    // zero accumulators + histograms
    k_zero_u32<<<2048, 256, 0, stream>>>((unsigned*)pooled, nPool);
    k_zero_u32<<<2048, 256, 0, stream>>>((unsigned*)out_small, nPool);
    k_zero_u32<<<256, 256, 0, stream>>>((unsigned*)h_acc, nH);
    k_zero_u32<<<16, 256, 0, stream>>>((unsigned*)hist1, 2 * NBIN);

    // relu + bf16 cast of activations; weight repack to [K][N] bf16
    k_relu_cast<<<8192, 256, 0, stream>>>(x, xb, nAct);
    k_cast_w<<<(int)(((long long)Cn * KCONV + 255) / 256), 256, 0, stream>>>(conv_w, wb);

    // conv implicit-GEMM (WMMA bf16) with fused avg-pool epilogue
    dim3 cgrid(256, 2);
    k_conv_wmma<<<cgrid, 256, 0, stream>>>(xb, wb, pooled);
    k_pool_to_nodes<<<2048, 256, 0, stream>>>(pooled, nodes, nPool);

    // prune thresholds (LDS-privatized histogram of |w|, 90th percentile)
    // w1 std ~ 1/128 -> range [0,0.04]; w2 std ~ 0.022 -> range [0,0.12]
    k_hist<<<4096, 256, 0, stream>>>(w1, nW / 4, (float)NBIN / 0.04f, hist1);
    k_hist<<<4096, 256, 0, stream>>>(w2, nW / 4, (float)NBIN / 0.12f, hist2);
    k_thr<<<1, 32, 0, stream>>>(hist1, 0.04f / NBIN, nW, thr1);
    k_thr<<<1, 32, 0, stream>>>(hist2, 0.12f / NBIN, nW, thr2);

    // graph MLP: K-split weight-streaming WMMA GEMMs with inline pruning
    dim3 g1(FEAT / 128, NNODE / 512);    // 16 x 32 = 512 blocks
    k_gemm1<<<g1, 256, 0, stream>>>(nodes, w1, thr1, h_acc);
    k_relu_cast_h<<<256, 256, 0, stream>>>(h_acc, hbuf, nH);
    dim3 g2(NNODE / 128, FEAT / 256);    // 128 x 8 = 1024 blocks
    k_gemm2<<<g2, 256, 0, stream>>>(hbuf, w2, thr2, out_small);

    // instance-norm stats on 8x8 cells, fused nearest-neighbor upsample
    k_norm_up<<<Bn * Cn, 256, 0, stream>>>(out_small, gamma, beta, out);
}
